// EquivariantBlock_41120016892395
// MI455X (gfx1250) — matrix operand order repaired and checked by reading, not compiled, not run on hardware
//
#include <hip/hip_runtime.h>
#include <hip/hip_bf16.h>

typedef __attribute__((ext_vector_type(16))) _Float16 v16h;
typedef __attribute__((ext_vector_type(8)))  float    v8f;

#define BATCH 4
#define NTOK  2048
#define SDIM  256
#define CDIM  64
#define NH    4
#define HD    64
#define BN    (BATCH * NTOK)
#define LN_EPS 1e-5f
#define TN 4  // N-tiles (of 16) per wave in GEMMs -> 16x64 per wave

// ---------------------------------------------------------------------------
// WMMA helpers (gfx1250 wave32, 16x16x32 f16 -> f32)
// ---------------------------------------------------------------------------
__device__ __forceinline__ v8f vzero8() {
  v8f z = {0.f, 0.f, 0.f, 0.f, 0.f, 0.f, 0.f, 0.f};
  return z;
}

__device__ __forceinline__ v8f wmma_f16f32(v16h a, v16h b, v8f c) {
  return __builtin_amdgcn_wmma_f32_16x16x32_f16(false, a, false, b, (short)0, c,
                                                false, false);
}

// A-matrix fragment: 16x32 f16 tile, row-major source with leading dim lda.
// Lane l (l2=l%16, hc=l/16) holds row M=l2; VGPR v (packed half-pairs):
//   v<4 : K = 8*hc + 2v,{+1}      v>=4: K = 16 + 8*hc + 2(v-4),{+1}
__device__ __forceinline__ v16h load_a_frag(const _Float16* A, int lda, int lane) {
  int l2 = lane & 15, hc = (lane >> 4) & 1;
  const _Float16* ar = A + (size_t)l2 * lda;
  union { v16h h; float f[8]; } u;
#pragma unroll
  for (int v = 0; v < 8; ++v) {
    int kk = (v < 4) ? (8 * hc + 2 * v) : (16 + 8 * hc + 2 * (v - 4));
    u.f[v] = *reinterpret_cast<const float*>(ar + kk);
  }
  return u.h;
}

// B-matrix fragment: 32x16 f16 tile (K x N), row-major source, leading dim ldb.
// Lane l holds B column N = l%16; element s: K = 8*(l/16) + s (s<8), 16 + 8*(l/16) + s-8 (s>=8).
__device__ __forceinline__ v16h load_b_frag(const _Float16* B, int ldb, int lane) {
  int l2 = lane & 15, hc = (lane >> 4) & 1;
  const _Float16* br = B + l2;
  v16h r;
#pragma unroll
  for (int s = 0; s < 16; ++s) {
    int kk = (s < 8) ? (8 * hc + s) : (16 + 8 * hc + (s - 8));
    r[s] = br[(size_t)kk * ldb];
  }
  return r;
}
// D/C tile element mapping: VGPR r, lane l -> M = r + 8*(l/16), N = l%16.

__device__ __forceinline__ float gelu_exact(float x) {
  return 0.5f * x * (1.0f + erff(x * 0.70710678118654752f));
}

__device__ __forceinline__ float softplus_f(float x) {
  return (x > 20.0f) ? x : log1pf(__expf(x));
}

// ---------------------------------------------------------------------------
// Async B-panel staging (gfx1250): copy 32 x 64 f16 panel (4 KiB) to LDS with
// global_load_async_to_lds_b128 (ASYNCcnt), double buffered, strength-reduced
// addressing (pointer adds only in the loop).
// ---------------------------------------------------------------------------
__device__ __forceinline__ void issue_b_panel(const char* gbase, size_t rowstep,
                                              uint32_t lbase) {
  const char* g = gbase;
  uint32_t l = lbase;
#pragma unroll
  for (int c = 0; c < 8; ++c) {
    asm volatile("global_load_async_to_lds_b128 %0, %1, off"
                 :: "v"(l), "v"(g) : "memory");
    g += rowstep;
    l += 512;
  }
}

// Mainloop: acc[0..3] += A[16 x K] @ B[K x 64]  (A at row0 applied, lda=K;
// Bp = B + col0, ldb given). Bs = shared, 2 buffers of 32*64 halves.
__device__ __forceinline__ void gemm_mainloop(const _Float16* __restrict__ A,
                                              const _Float16* __restrict__ Bp,
                                              int K, int ldb, int lane,
                                              _Float16* Bs, v8f acc[TN]) {
  uint32_t lds0 = (uint32_t)(uintptr_t)Bs;
  const uint32_t bufbytes = 32 * 64 * 2;
  const size_t rowstep = (size_t)ldb * 8;     // 4 rows * ldb * 2B
  const size_t panelstep = (size_t)ldb * 64;  // 32 rows * ldb * 2B
  // per-lane source base: row = lane>>3, 16B chunk = (lane&7)*16
  const char* gbase =
      (const char*)(Bp + (size_t)(lane >> 3) * ldb) + (lane & 7) * 16;
  uint32_t lbase = lds0 + (uint32_t)(lane * 16);

  issue_b_panel(gbase, rowstep, lbase);  // panel 0 -> buf 0
  gbase += panelstep;

  int nk = K / 32;
  // steady state: issue next panel, wait for current, compute
  for (int i = 0; i < nk - 1; ++i) {
    int buf = i & 1;
    issue_b_panel(gbase, rowstep, lbase + (uint32_t)(buf ^ 1) * bufbytes);
    gbase += panelstep;
    asm volatile("s_wait_asynccnt 0x8" ::: "memory");
    v16h a = load_a_frag(A + i * 32, K, lane);
    const _Float16* Bl = Bs + buf * (32 * 64);
#pragma unroll
    for (int j = 0; j < TN; ++j) {
      v16h b = load_b_frag(Bl + j * 16, 64, lane);
      acc[j] = wmma_f16f32(a, b, acc[j]);
    }
  }
  // peeled last iteration
  {
    int i = nk - 1;
    asm volatile("s_wait_asynccnt 0x0" ::: "memory");
    v16h a = load_a_frag(A + i * 32, K, lane);
    const _Float16* Bl = Bs + (i & 1) * (32 * 64);
#pragma unroll
    for (int j = 0; j < TN; ++j) {
      v16h b = load_b_frag(Bl + j * 16, 64, lane);
      acc[j] = wmma_f16f32(a, b, acc[j]);
    }
  }
}

// ---------------------------------------------------------------------------
// Weight conversion f32 -> f16
// ---------------------------------------------------------------------------
__global__ void k_f32_to_f16(const float* __restrict__ src,
                             _Float16* __restrict__ dst, int n) {
  int i = blockIdx.x * blockDim.x + threadIdx.x;
  if (i < n) dst[i] = (_Float16)src[i];
}

// ---------------------------------------------------------------------------
// Equivariant LayerNorm: scalar LN over S plus vector norm rescale.
// vn16 is written transposed: vn16[(bn*3 + x)*CDIM + c]
// ---------------------------------------------------------------------------
__global__ __launch_bounds__(256) void k_eq_ln(
    const float* __restrict__ s, const float* __restrict__ v,
    const float* __restrict__ g, const float* __restrict__ be,
    const float* __restrict__ vscale, _Float16* __restrict__ sn16,
    _Float16* __restrict__ vn16) {
  __shared__ float red[256];
  __shared__ float vnorm[CDIM];
  __shared__ float stats[2];
  int bn = blockIdx.x;
  int t = threadIdx.x;

  float x = s[(size_t)bn * SDIM + t];
  red[t] = x;
  __syncthreads();
  for (int st = 128; st > 0; st >>= 1) {
    if (t < st) red[t] += red[t + st];
    __syncthreads();
  }
  if (t == 0) stats[0] = red[0] * (1.0f / SDIM);
  __syncthreads();
  float mu = stats[0];
  float dx = x - mu;
  red[t] = dx * dx;
  __syncthreads();
  for (int st = 128; st > 0; st >>= 1) {
    if (t < st) red[t] += red[t + st];
    __syncthreads();
  }
  if (t == 0) stats[1] = rsqrtf(red[0] * (1.0f / SDIM) + LN_EPS);
  __syncthreads();
  float rstd = stats[1];
  sn16[(size_t)bn * SDIM + t] = (_Float16)(dx * rstd * g[t] + be[t]);

  if (t < CDIM) {
    const float* vp = v + ((size_t)bn * CDIM + t) * 3;
    float a0 = vp[0], a1 = vp[1], a2 = vp[2];
    vnorm[t] = sqrtf(a0 * a0 + a1 * a1 + a2 * a2);
  }
  __syncthreads();
  if (t < CDIM) red[t] = vnorm[t];
  __syncthreads();
  for (int st = 32; st > 0; st >>= 1) {
    if (t < st) red[t] += red[t + st];
    __syncthreads();
  }
  float meanv = red[0] * (1.0f / CDIM);
  if (t < 3 * CDIM) {
    int xc = t >> 6;  // 0..2
    int c = t & 63;
    float val = v[((size_t)bn * CDIM + c) * 3 + xc] / (vnorm[c] + LN_EPS) *
                meanv * vscale[c];
    vn16[((size_t)bn * 3 + xc) * CDIM + c] = (_Float16)val;
  }
}

// ---------------------------------------------------------------------------
// GEMM kernels: one wave computes 16 rows x 64 cols (4 WMMA tiles, A reuse),
// B panels staged through LDS with async copies.
// ---------------------------------------------------------------------------
__global__ __launch_bounds__(32) void k_gemm_f16(
    const _Float16* __restrict__ A, const _Float16* __restrict__ B,
    const float* __restrict__ bias, _Float16* __restrict__ Out, int M, int Nc,
    int K, int act) {
  __shared__ __align__(16) _Float16 Bs[2][32 * 64];
  int lane = threadIdx.x & 31;
  int col0 = blockIdx.x * 64;
  int row0 = blockIdx.y * 16;
  int l2 = lane & 15, hc = (lane >> 4) & 1;
  v8f acc[TN];
#pragma unroll
  for (int j = 0; j < TN; ++j) acc[j] = vzero8();
  gemm_mainloop(A + (size_t)row0 * K, B + col0, K, Nc, lane, &Bs[0][0], acc);
#pragma unroll
  for (int j = 0; j < TN; ++j) {
    int col = col0 + j * 16 + l2;
    float bv = bias ? bias[col] : 0.0f;
#pragma unroll
    for (int r = 0; r < 8; ++r) {
      float val = acc[j][r] + bv;
      if (act == 1) val = gelu_exact(val);
      Out[(size_t)(row0 + r + 8 * hc) * Nc + col] = (_Float16)val;
    }
  }
}

__global__ __launch_bounds__(32) void k_gemm_f32res(
    const _Float16* __restrict__ A, const _Float16* __restrict__ B,
    const float* __restrict__ bias, const float* __restrict__ Res,
    float* __restrict__ Out, int M, int Nc, int K) {
  __shared__ __align__(16) _Float16 Bs[2][32 * 64];
  int lane = threadIdx.x & 31;
  int col0 = blockIdx.x * 64;
  int row0 = blockIdx.y * 16;
  int l2 = lane & 15, hc = (lane >> 4) & 1;
  v8f acc[TN];
#pragma unroll
  for (int j = 0; j < TN; ++j) acc[j] = vzero8();
  gemm_mainloop(A + (size_t)row0 * K, B + col0, K, Nc, lane, &Bs[0][0], acc);
#pragma unroll
  for (int j = 0; j < TN; ++j) {
    int col = col0 + j * 16 + l2;
    float bv = bias ? bias[col] : 0.0f;
#pragma unroll
    for (int r = 0; r < 8; ++r) {
      size_t idx = (size_t)(row0 + r + 8 * hc) * Nc + col;
      Out[idx] = Res[idx] + acc[j][r] + bv;
    }
  }
}

// Vector-channel GEMM with residual into (B,N,C,3) layout.
// Rows are (bn,x) pairs; output element (row,col) -> Out[(bn*CDIM+col)*3 + x].
__global__ __launch_bounds__(32) void k_gemm_vres(
    const _Float16* __restrict__ A, const _Float16* __restrict__ B,
    const float* __restrict__ Res, float* __restrict__ Out, int M, int Nc,
    int K) {
  __shared__ __align__(16) _Float16 Bs[2][32 * 64];
  int lane = threadIdx.x & 31;
  int col0 = blockIdx.x * 64;
  int row0 = blockIdx.y * 16;
  int l2 = lane & 15, hc = (lane >> 4) & 1;
  v8f acc[TN];
#pragma unroll
  for (int j = 0; j < TN; ++j) acc[j] = vzero8();
  gemm_mainloop(A + (size_t)row0 * K, B + col0, K, Nc, lane, &Bs[0][0], acc);
#pragma unroll
  for (int j = 0; j < TN; ++j) {
    int col = col0 + j * 16 + l2;
#pragma unroll
    for (int r = 0; r < 8; ++r) {
      int row = row0 + r + 8 * hc;
      int bn = row / 3, x = row - bn * 3;
      size_t idx = ((size_t)bn * CDIM + col) * 3 + x;
      Out[idx] = Res[idx] + acc[j][r];
    }
  }
}

// QKV projection with per-head scatter. mode 0/2: [b][h][n][d]; mode 1: K^T [b][h][d][n]
__global__ __launch_bounds__(32) void k_gemm_qkv(
    const _Float16* __restrict__ A, const _Float16* __restrict__ W,
    const float* __restrict__ bias, _Float16* __restrict__ Out, int mode) {
  __shared__ __align__(16) _Float16 Bs[2][32 * 64];
  int lane = threadIdx.x & 31;
  int col0 = blockIdx.x * 64;
  int row0 = blockIdx.y * 16;
  int l2 = lane & 15, hc = (lane >> 4) & 1;
  v8f acc[TN];
#pragma unroll
  for (int j = 0; j < TN; ++j) acc[j] = vzero8();
  gemm_mainloop(A + (size_t)row0 * SDIM, W + col0, SDIM, SDIM, lane, &Bs[0][0],
                acc);
#pragma unroll
  for (int j = 0; j < TN; ++j) {
    int col = col0 + j * 16 + l2;
    int h = col >> 6, d = col & 63;
    float bv = bias[col];
#pragma unroll
    for (int r = 0; r < 8; ++r) {
      int row = row0 + r + 8 * hc;
      int b_ = row >> 11, n = row & (NTOK - 1);
      float val = acc[j][r] + bv;
      size_t idx;
      if (mode == 1)
        idx = ((size_t)(b_ * NH + h) * HD + d) * NTOK + n;  // K transposed
      else
        idx = ((size_t)(b_ * NH + h) * NTOK + n) * HD + d;
      Out[idx] = (_Float16)val;
    }
  }
}

// ---------------------------------------------------------------------------
// Fused attention: per (16-query tile, batch) wave. Two-pass softmax with
// per-lane online stats (single cross-lane merge after pass 1); WMMA for
// QK^T, P@V (per head) and P@v_mix (summed over heads -> /H == attn_mean).
// ---------------------------------------------------------------------------
__global__ __launch_bounds__(32, 1) void k_attn(
    const _Float16* __restrict__ qh, const _Float16* __restrict__ kt,
    const _Float16* __restrict__ vh, const _Float16* __restrict__ vmix,
    const float* __restrict__ pos, const float* __restrict__ w_dist,
    const float* __restrict__ b_dist, _Float16* __restrict__ as16,
    _Float16* __restrict__ av16) {
  __shared__ __align__(16) _Float16 pP[16 * 32];
  __shared__ float qpos[16][3];
  int lane = threadIdx.x & 31;
  int l2 = lane & 15, hc = (lane >> 4) & 1;
  int b = blockIdx.y;
  int row0 = blockIdx.x * 16;

  if (lane < 16) {
    const float* pp = pos + ((size_t)b * NTOK + row0 + lane) * 3;
    qpos[lane][0] = pp[0];
    qpos[lane][1] = pp[1];
    qpos[lane][2] = pp[2];
  }
  __syncthreads();

  v8f vacc[12];
#pragma unroll
  for (int j = 0; j < 12; ++j) vacc[j] = vzero8();

  const _Float16* vmixb = vmix + (size_t)b * NTOK * 192;

  for (int h = 0; h < NH; ++h) {
    const _Float16* Q = qh + (size_t)(b * NH + h) * NTOK * HD;
    const _Float16* KT = kt + (size_t)(b * NH + h) * HD * NTOK;
    const _Float16* V = vh + (size_t)(b * NH + h) * NTOK * HD;
    v16h aq0 = load_a_frag(Q + (size_t)row0 * HD + 0, HD, lane);
    v16h aq1 = load_a_frag(Q + (size_t)row0 * HD + 32, HD, lane);
    float wd = w_dist[h], bd = b_dist[h];

    float mrow[8], lrow[8];
#pragma unroll
    for (int r = 0; r < 8; ++r) {
      mrow[r] = -1e30f;
      lrow[r] = 0.0f;
    }

    // ---- pass 1: per-lane online max/sum over this lane's columns ----
    for (int m0 = 0; m0 < NTOK; m0 += 16) {
      v16h bk0 = load_b_frag(KT + m0, NTOK, lane);
      v16h bk1 = load_b_frag(KT + (size_t)32 * NTOK + m0, NTOK, lane);
      v8f sc = vzero8();
      sc = wmma_f16f32(aq0, bk0, sc);
      sc = wmma_f16f32(aq1, bk1, sc);
      const float* kp = pos + ((size_t)b * NTOK + m0 + l2) * 3;
      float kx = kp[0], ky = kp[1], kz = kp[2];
#pragma unroll
      for (int r = 0; r < 8; ++r) {
        int qr = r + 8 * hc;
        float dx = qpos[qr][0] - kx, dy = qpos[qr][1] - ky,
              dz = qpos[qr][2] - kz;
        float dist = sqrtf(fmaxf(dx * dx + dy * dy + dz * dz, 1e-12f));
        float bias = softplus_f(dist * wd + bd);
        float sv = sc[r] * 0.125f - bias;  // 1/sqrt(64)
        float mo = mrow[r];
        float mn = fmaxf(mo, sv);
        lrow[r] = lrow[r] * __expf(mo - mn) + __expf(sv - mn);
        mrow[r] = mn;
      }
    }
    // merge per-lane stats across the 16 lanes of each half-wave
#pragma unroll
    for (int r = 0; r < 8; ++r) {
#pragma unroll
      for (int msk = 1; msk < 16; msk <<= 1) {
        float om = __shfl_xor(mrow[r], msk, 32);
        float os = __shfl_xor(lrow[r], msk, 32);
        float mn = fmaxf(mrow[r], om);
        lrow[r] = lrow[r] * __expf(mrow[r] - mn) + os * __expf(om - mn);
        mrow[r] = mn;
      }
      lrow[r] = 1.0f / lrow[r];
    }

    v8f oacc[4];
#pragma unroll
    for (int j = 0; j < 4; ++j) oacc[j] = vzero8();

    // ---- pass 2: normalized P, then P@V and P@v_mix ----
    for (int m0 = 0; m0 < NTOK; m0 += 32) {
#pragma unroll
      for (int sub = 0; sub < 2; ++sub) {
        int mm = m0 + sub * 16;
        v16h bk0 = load_b_frag(KT + mm, NTOK, lane);
        v16h bk1 = load_b_frag(KT + (size_t)32 * NTOK + mm, NTOK, lane);
        v8f sc = vzero8();
        sc = wmma_f16f32(aq0, bk0, sc);
        sc = wmma_f16f32(aq1, bk1, sc);
        const float* kp = pos + ((size_t)b * NTOK + mm + l2) * 3;
        float kx = kp[0], ky = kp[1], kz = kp[2];
#pragma unroll
        for (int r = 0; r < 8; ++r) {
          int qr = r + 8 * hc;
          float dx = qpos[qr][0] - kx, dy = qpos[qr][1] - ky,
                dz = qpos[qr][2] - kz;
          float dist = sqrtf(fmaxf(dx * dx + dy * dy + dz * dz, 1e-12f));
          float bias = softplus_f(dist * wd + bd);
          float sv = sc[r] * 0.125f - bias;
          float p = __expf(sv - mrow[r]) * lrow[r];
          pP[qr * 32 + sub * 16 + l2] = (_Float16)p;
        }
      }
      __syncthreads();
      v16h pa = load_a_frag(pP, 32, lane);
#pragma unroll
      for (int j = 0; j < 4; ++j) {
        v16h bv = load_b_frag(V + (size_t)m0 * HD + j * 16, HD, lane);
        oacc[j] = wmma_f16f32(pa, bv, oacc[j]);
      }
#pragma unroll
      for (int j = 0; j < 12; ++j) {
        v16h bm = load_b_frag(vmixb + (size_t)m0 * 192 + j * 16, 192, lane);
        vacc[j] = wmma_f16f32(pa, bm, vacc[j]);
      }
      __syncthreads();
    }

    // store per-head attention output (pre-Wo), concat layout [bn][h*64+d]
#pragma unroll
    for (int j = 0; j < 4; ++j)
#pragma unroll
      for (int r = 0; r < 8; ++r) {
        int qr = r + 8 * hc;
        as16[(size_t)(b * NTOK + row0 + qr) * SDIM + h * HD + j * 16 + l2] =
            (_Float16)oacc[j][r];
      }
  }

  // head-mean vector mix output (pre-Wvo), layout [(bn*3+x)*64+e]
#pragma unroll
  for (int j = 0; j < 12; ++j)
#pragma unroll
    for (int r = 0; r < 8; ++r) {
      int qr = r + 8 * hc;
      av16[(size_t)(b * NTOK + row0 + qr) * 192 + j * 16 + l2] =
          (_Float16)(vacc[j][r] * 0.25f);
    }
}

// ---------------------------------------------------------------------------
// Host launcher
// ---------------------------------------------------------------------------
extern "C" void kernel_launch(void* const* d_in, const int* in_sizes, int n_in,
                              void* d_out, int out_size, void* d_ws,
                              size_t ws_size, hipStream_t stream) {
  (void)in_sizes; (void)n_in; (void)out_size; (void)ws_size;
  const float* s_in = (const float*)d_in[0];
  const float* v_in = (const float*)d_in[1];
  const float* pos = (const float*)d_in[2];
  const float* Wq = (const float*)d_in[3];
  const float* bq = (const float*)d_in[4];
  const float* Wk = (const float*)d_in[5];
  const float* bk = (const float*)d_in[6];
  const float* Wv = (const float*)d_in[7];
  const float* bv = (const float*)d_in[8];
  const float* Wo = (const float*)d_in[9];
  const float* bo = (const float*)d_in[10];
  const float* w_dist = (const float*)d_in[11];
  const float* b_dist = (const float*)d_in[12];
  const float* Wvv = (const float*)d_in[13];
  const float* Wvo = (const float*)d_in[14];
  const float* g1 = (const float*)d_in[15];
  const float* be1 = (const float*)d_in[16];
  const float* vs1 = (const float*)d_in[17];
  const float* g2 = (const float*)d_in[18];
  const float* be2 = (const float*)d_in[19];
  const float* vs2 = (const float*)d_in[20];
  const float* Wf1 = (const float*)d_in[21];
  const float* bf1 = (const float*)d_in[22];
  const float* Wf2 = (const float*)d_in[23];
  const float* bf2 = (const float*)d_in[24];
  const float* Wfv1 = (const float*)d_in[25];
  const float* Wfv2 = (const float*)d_in[26];

  float* out_s = (float*)d_out;                  // (B,N,S)
  float* out_v = out_s + (size_t)BN * SDIM;      // (B,N,C,3)

  char* ws = (char*)d_ws;
  size_t off = 0;
  auto take = [&](size_t bytes) -> void* {
    void* p = ws + off;
    off = (off + bytes + 255) & ~(size_t)255;
    return p;
  };
  _Float16* sn16 = (_Float16*)take((size_t)BN * SDIM * 2);
  _Float16* q16 = (_Float16*)take((size_t)BN * SDIM * 2);
  _Float16* k16 = (_Float16*)take((size_t)BN * SDIM * 2);   // transposed per head
  _Float16* vv16 = (_Float16*)take((size_t)BN * SDIM * 2);
  _Float16* vmix16 = (_Float16*)take((size_t)BN * 192 * 2);
  _Float16* as16 = (_Float16*)take((size_t)BN * SDIM * 2);
  _Float16* av16 = (_Float16*)take((size_t)BN * 192 * 2);
  _Float16* h116 = (_Float16*)take((size_t)BN * 1024 * 2);  // reused for fv1
  _Float16* vn16 = (_Float16*)take((size_t)BN * 192 * 2);
  _Float16* Wq16 = (_Float16*)take(65536 * 2);
  _Float16* Wk16 = (_Float16*)take(65536 * 2);
  _Float16* Wv16 = (_Float16*)take(65536 * 2);
  _Float16* Wo16 = (_Float16*)take(65536 * 2);
  _Float16* Wvv16 = (_Float16*)take(4096 * 2);
  _Float16* Wvo16 = (_Float16*)take(4096 * 2);
  _Float16* Wf116 = (_Float16*)take(262144 * 2);
  _Float16* Wf216 = (_Float16*)take(262144 * 2);
  _Float16* Wfv116 = (_Float16*)take(16384 * 2);
  _Float16* Wfv216 = (_Float16*)take(16384 * 2);

  auto cvt = [&](const float* src, _Float16* dst, int n) {
    k_f32_to_f16<<<(n + 255) / 256, 256, 0, stream>>>(src, dst, n);
  };
  cvt(Wq, Wq16, 65536);
  cvt(Wk, Wk16, 65536);
  cvt(Wv, Wv16, 65536);
  cvt(Wo, Wo16, 65536);
  cvt(Wvv, Wvv16, 4096);
  cvt(Wvo, Wvo16, 4096);
  cvt(Wf1, Wf116, 262144);
  cvt(Wf2, Wf216, 262144);
  cvt(Wfv1, Wfv116, 16384);
  cvt(Wfv2, Wfv216, 16384);

  // --- norm1 ---
  k_eq_ln<<<BN, 256, 0, stream>>>(s_in, v_in, g1, be1, vs1, sn16, vn16);

  // --- QKV projections (per-head layouts) ---
  dim3 gproj(SDIM / 64, BN / 16);
  k_gemm_qkv<<<gproj, 32, 0, stream>>>(sn16, Wq16, bq, q16, 0);
  k_gemm_qkv<<<gproj, 32, 0, stream>>>(sn16, Wk16, bk, k16, 1);
  k_gemm_qkv<<<gproj, 32, 0, stream>>>(sn16, Wv16, bv, vv16, 2);

  // --- v_mix = vn^T @ Wvv  (rows = (bn,x)) ---
  k_gemm_f16<<<dim3(CDIM / 64, BN * 3 / 16), 32, 0, stream>>>(
      vn16, Wvv16, nullptr, vmix16, BN * 3, CDIM, CDIM, 0);

  // --- fused attention ---
  k_attn<<<dim3(NTOK / 16, BATCH), 32, 0, stream>>>(
      q16, k16, vv16, vmix16, pos, w_dist, b_dist, as16, av16);

  // --- s = s + attn @ Wo + bo ---
  k_gemm_f32res<<<dim3(SDIM / 64, BN / 16), 32, 0, stream>>>(
      as16, Wo16, bo, s_in, out_s, BN, SDIM, SDIM);

  // --- v = v + v_attn @ Wvo ---
  k_gemm_vres<<<dim3(CDIM / 64, BN * 3 / 16), 32, 0, stream>>>(
      av16, Wvo16, v_in, out_v, BN * 3, CDIM, CDIM);

  // --- norm2 ---
  k_eq_ln<<<BN, 256, 0, stream>>>(out_s, out_v, g2, be2, vs2, sn16, vn16);

  // --- FFN: h1 = gelu(sn2 @ Wf1 + bf1) ; s += h1 @ Wf2 + bf2 ---
  k_gemm_f16<<<dim3(1024 / 64, BN / 16), 32, 0, stream>>>(
      sn16, Wf116, bf1, h116, BN, 1024, SDIM, 1);
  k_gemm_f32res<<<dim3(SDIM / 64, BN / 16), 32, 0, stream>>>(
      h116, Wf216, bf2, out_s, out_s, BN, SDIM, 1024);

  // --- vector FFN: t = vn2^T @ Wfv1 ; v += t @ Wfv2 ---
  k_gemm_f16<<<dim3(256 / 64, BN * 3 / 16), 32, 0, stream>>>(
      vn16, Wfv116, nullptr, h116, BN * 3, 256, CDIM, 0);
  k_gemm_vres<<<dim3(CDIM / 64, BN * 3 / 16), 32, 0, stream>>>(
      h116, Wfv216, out_v, out_v, BN * 3, CDIM, 256);
}
